// AATA_81209241633306
// MI455X (gfx1250) — matrix operand order repaired and checked by reading, not compiled, not run on hardware
//
#include <hip/hip_runtime.h>
#include <hip/hip_bf16.h>

typedef __attribute__((ext_vector_type(16))) _Float16 v16h;
typedef __attribute__((ext_vector_type(8)))  _Float16 v8h;
typedef __attribute__((ext_vector_type(4)))  _Float16 v4h;
typedef __attribute__((ext_vector_type(8)))  float    v8f;
typedef __attribute__((ext_vector_type(4)))  float    v4f;

#define BB 128
#define TT 2048
#define HH 128

// Hardware tanh (v_tanh_f32, TRANS32 op on gfx1250) when the builtin exists;
// guarded so the source compiles on any toolchain.
__device__ __forceinline__ float fast_tanh(float x) {
#if __has_builtin(__builtin_amdgcn_tanhf)
  return __builtin_amdgcn_tanhf(x);
#elif __has_builtin(__builtin_amdgcn_tanh_f32)
  return __builtin_amdgcn_tanh_f32(x);
#else
  return tanhf(x);
#endif
}

// ---------------------------------------------------------------------------
// A-fragment: 16x32 f16, built from fp32 global rows.
// ISA layout (16-bit A 16x32): lanes 0-15 -> M=lane, elements 0..7 = K 0..7,
// elements 8..15 = K 16..23; lanes 16-31 -> M=lane-16, K 8..15 / 24..31.
// koff = (lane<16)?0:8.
// ---------------------------------------------------------------------------
__device__ __forceinline__ v16h make_a_frag(const float* __restrict__ row,
                                            int kbase, int koff) {
  const v4f* p0 = (const v4f*)(row + kbase + koff);
  v4f x0 = p0[0], x1 = p0[1];
  const v4f* p1 = (const v4f*)(row + kbase + koff + 16);
  v4f y0 = p1[0], y1 = p1[1];
  v16h a;
#pragma unroll
  for (int i = 0; i < 4; ++i) {
    a[i]      = (_Float16)x0[i];
    a[4 + i]  = (_Float16)x1[i];
    a[8 + i]  = (_Float16)y0[i];
    a[12 + i] = (_Float16)y1[i];
  }
  return a;
}

// ---------------------------------------------------------------------------
// B-fragment: 32x16 f16 from LDS. ISA layout (16-bit B): lane&15 = N column,
// lanes 0-15 hold K = 0..7,16..23, lanes 16-31 hold K = 8..15,24..31
// (kstart includes the 8-lane-half offset) -> 2x ds_load_b128 per lane.
// ---------------------------------------------------------------------------
__device__ __forceinline__ v16h make_b_frag(const _Float16* __restrict__ wrow,
                                            int kstart) {
  const v8h* p = (const v8h*)(wrow + kstart);
  v8h lo = p[0], hi = p[2];
  return __builtin_shufflevector(lo, hi, 0, 1, 2, 3, 4, 5, 6, 7,
                                 8, 9, 10, 11, 12, 13, 14, 15);
}

// ---------------------------------------------------------------------------
// Kernel 1: energy[b*T+t] = W_a . tanh(W_h x_pos + W_d x_dyn) + b_a
// 256 threads = 8 wave32; each wave computes a 16-row strip x all 128 outputs
// with v_wmma_f32_16x16x32_f16, then fuses tanh + GEMV epilogue.
// ---------------------------------------------------------------------------
__global__ __launch_bounds__(256) void aata_energy_kernel(
    const float* __restrict__ Hpos, const float* __restrict__ Hdyn,
    const float* __restrict__ Wh, const float* __restrict__ Wd,
    const float* __restrict__ Wa, const float* __restrict__ ba_p,
    float* __restrict__ energy) {
  __shared__ __align__(16) _Float16 sWh[HH * HH];
  __shared__ __align__(16) _Float16 sWd[HH * HH];

  const int tid = threadIdx.x;

  // Stage both weight matrices into LDS as f16 (row-major [o][h]).
  for (int i = tid * 4; i < HH * HH; i += 256 * 4) {
    v4f wh = *(const v4f*)(Wh + i);
    v4f wd = *(const v4f*)(Wd + i);
    v4h hh, hd;
#pragma unroll
    for (int j = 0; j < 4; ++j) {
      hh[j] = (_Float16)wh[j];
      hd[j] = (_Float16)wd[j];
    }
    *(v4h*)(sWh + i) = hh;
    *(v4h*)(sWd + i) = hd;
  }
  __syncthreads();

  const int  lane = tid & 31;
  const int  wave = tid >> 5;
  const int  lrow = lane & 15;
  const int  koff = (lane < 16) ? 0 : 8;   // A-fragment K offset
  const int  kb2  = (lane < 16) ? 0 : 8;   // B-fragment K offset
  const long strip = (long)blockIdx.x * 8 + wave;
  const long row0  = strip * 16;           // flattened (b*T + t) row base

  const float* prow_pos = Hpos + (row0 + lrow) * HH;
  const float* prow_dyn = Hdyn + (row0 + lrow) * HH;

  // Preload all A fragments (K = 128 -> 4 steps of 32) for both inputs.
  v16h ap[4], ad[4];
#pragma unroll
  for (int ks = 0; ks < 4; ++ks) {
    ap[ks] = make_a_frag(prow_pos, ks * 32, koff);
    ad[ks] = make_a_frag(prow_dyn, ks * 32, koff);
  }

  // W_a value for this lane's output column in each N-tile.
  float wa[8];
#pragma unroll
  for (int nt = 0; nt < 8; ++nt) wa[nt] = Wa[nt * 16 + lrow];

  float e[8];
#pragma unroll
  for (int r = 0; r < 8; ++r) e[r] = 0.f;

#pragma unroll
  for (int nt = 0; nt < 8; ++nt) {
    v8f c = {};
    const _Float16* wrh = sWh + (nt * 16 + lrow) * HH;
    const _Float16* wrd = sWd + (nt * 16 + lrow) * HH;
#pragma unroll
    for (int ks = 0; ks < 4; ++ks) {
      v16h bh = make_b_frag(wrh, ks * 32 + kb2);
      c = __builtin_amdgcn_wmma_f32_16x16x32_f16(false, ap[ks], false, bh,
                                                 (short)0, c, false, false);
      v16h bd = make_b_frag(wrd, ks * 32 + kb2);
      c = __builtin_amdgcn_wmma_f32_16x16x32_f16(false, ad[ks], false, bd,
                                                 (short)0, c, false, false);
    }
    // C/D layout: lane&15 = N column; VGPR r = row M=r (lanes<16) or M=r+8.
#pragma unroll
    for (int r = 0; r < 8; ++r) e[r] += fast_tanh(c[r]) * wa[nt];
  }

  // Butterfly-reduce over the 16 lanes sharing each row (stays within halves).
#pragma unroll
  for (int m = 1; m < 16; m <<= 1) {
#pragma unroll
    for (int r = 0; r < 8; ++r) e[r] += __shfl_xor(e[r], m, 32);
  }

  if (lrow == 0) {
    const float ba   = *ba_p;
    const long  base = row0 + ((lane < 16) ? 0 : 8);
#pragma unroll
    for (int r = 0; r < 8; ++r) energy[base + r] = e[r] + ba;
  }
}

// ---------------------------------------------------------------------------
// Kernel 2: per-batch softmax with acceleration scaling (energy -> alpha,
// in place inside d_out's alpha region). One block per batch, 256 threads,
// 8 T-elements per thread.
// ---------------------------------------------------------------------------
__global__ __launch_bounds__(256) void aata_alpha_kernel(
    const float* __restrict__ acc_w, const float* __restrict__ beta_p,
    float* __restrict__ ealpha) {
  const int b = blockIdx.x, tid = threadIdx.x;
  __shared__ float red[256];

  const float* aw = acc_w + (long)b * TT;
  float*       ea = ealpha + (long)b * TT;

  const float beta = *beta_p;
  const float bpos = fmaxf(beta, 0.f) + log1pf(expf(-fabsf(beta)));  // softplus

  float a_[8], s_[8];
  float amax = -1e30f;
#pragma unroll
  for (int i = 0; i < 8; ++i) {
    a_[i] = aw[tid + i * 256];
    amax  = fmaxf(amax, a_[i]);
  }
  red[tid] = amax;
  __syncthreads();
  for (int s = 128; s > 0; s >>= 1) {
    if (tid < s) red[tid] = fmaxf(red[tid], red[tid + s]);
    __syncthreads();
  }
  amax = fmaxf(red[0], 1e-6f);
  __syncthreads();

  float smax = -1e30f;
#pragma unroll
  for (int i = 0; i < 8; ++i) {
    const float scale = 1.f + bpos * (a_[i] / amax);
    s_[i] = ea[tid + i * 256] * scale;
    smax  = fmaxf(smax, s_[i]);
  }
  red[tid] = smax;
  __syncthreads();
  for (int s = 128; s > 0; s >>= 1) {
    if (tid < s) red[tid] = fmaxf(red[tid], red[tid + s]);
    __syncthreads();
  }
  smax = red[0];
  __syncthreads();

  float ssum = 0.f;
#pragma unroll
  for (int i = 0; i < 8; ++i) {
    s_[i] = expf(s_[i] - smax);
    ssum += s_[i];
  }
  red[tid] = ssum;
  __syncthreads();
  for (int s = 128; s > 0; s >>= 1) {
    if (tid < s) red[tid] += red[tid + s];
    __syncthreads();
  }
  const float inv = 1.f / red[0];
#pragma unroll
  for (int i = 0; i < 8; ++i) ea[tid + i * 256] = s_[i] * inv;
}

// ---------------------------------------------------------------------------
// Kernel 3a: zero the context slab (d_out is poisoned by the harness).
// ---------------------------------------------------------------------------
__global__ void aata_zero_ctx(float* __restrict__ ctx) {
  const int i = blockIdx.x * 256 + threadIdx.x;
  if (i < BB * HH) ctx[i] = 0.f;
}

// ---------------------------------------------------------------------------
// Kernel 3b: context[b,h] = sum_t alpha[b,t] * H_pos[b,t,h].
// Grid (B, 8 T-chunks), 128 threads (one per h); atomicAdd partials.
// ---------------------------------------------------------------------------
__global__ __launch_bounds__(128) void aata_context_kernel(
    const float* __restrict__ Hpos, const float* __restrict__ alpha,
    float* __restrict__ ctx) {
  const int b  = blockIdx.x;
  const int ch = blockIdx.y;
  const int h  = threadIdx.x;
  const int t0 = ch * (TT / 8);

  const float* al = alpha + (long)b * TT + t0;
  const float* hp = Hpos + ((long)b * TT + t0) * HH + h;

  float s = 0.f;
  for (int t = 0; t < TT / 8; ++t) s = fmaf(al[t], hp[(long)t * HH], s);
  atomicAdd(&ctx[b * HH + h], s);
}

extern "C" void kernel_launch(void* const* d_in, const int* in_sizes, int n_in,
                              void* d_out, int out_size, void* d_ws,
                              size_t ws_size, hipStream_t stream) {
  (void)in_sizes; (void)n_in; (void)out_size; (void)d_ws; (void)ws_size;

  const float* Hpos = (const float*)d_in[0];
  const float* Hdyn = (const float*)d_in[1];
  const float* accw = (const float*)d_in[2];
  const float* Wh   = (const float*)d_in[3];
  const float* Wd   = (const float*)d_in[4];
  const float* Wa   = (const float*)d_in[5];
  const float* ba   = (const float*)d_in[6];
  const float* beta = (const float*)d_in[7];

  float* ctx   = (float*)d_out;       // [B, H]
  float* alpha = ctx + BB * HH;       // [B, T] (energy staged here first)

  // 262144 rows / (16 rows/wave * 8 waves/block) = 2048 blocks.
  aata_energy_kernel<<<2048, 256, 0, stream>>>(Hpos, Hdyn, Wh, Wd, Wa, ba,
                                               alpha);
  aata_zero_ctx<<<(BB * HH + 255) / 256, 256, 0, stream>>>(ctx);
  aata_alpha_kernel<<<BB, 256, 0, stream>>>(accw, beta, alpha);
  aata_context_kernel<<<dim3(BB, 8), 128, 0, stream>>>(Hpos, alpha, ctx);
}
